// LigerLMHeadCE_19765439496673
// MI455X (gfx1250) — matrix operand order repaired and checked by reading, not compiled, not run on hardware
//
#include <hip/hip_runtime.h>
#include <hip/hip_bf16.h>
#include <math.h>

// ---------------------------------------------------------------------------
// Fused LM-head cross entropy for MI455X (gfx1250, wave32, WMMA).
//   logits = x @ W^T   (never materialized)
//   loss   = mean over valid rows of  [logsumexp(logits[n,:]) - logits[n,y[n]]]
// bf16 WMMA (f32 accum) tile GEMM. W tiles stream through a 4-deep circular
// LDS pipeline fed by CDNA5 GLOBAL_LOAD_ASYNC_TO_LDS_B128 (ASYNCcnt, in-order
// completion -> s_wait_asynccnt 2 releases the oldest tile). A fragments are
// software-pipelined from L2-resident x. Per-(row, V-block) softmax partials
// merged by online log-sum-exp.
// ---------------------------------------------------------------------------

typedef __attribute__((ext_vector_type(16))) __bf16 v16bf;
typedef __attribute__((ext_vector_type(8)))  __bf16 v8bf;
typedef __attribute__((ext_vector_type(4)))  __bf16 v4bf;
typedef __attribute__((ext_vector_type(8)))  float  v8f;

// Types for the async-to-LDS builtin: (v4i as1*, v4i as3*, imm, imm).
typedef int v4i __attribute__((vector_size(16)));
typedef __attribute__((address_space(1))) v4i* v4i_as1p;
typedef __attribute__((address_space(3))) v4i* v4i_as3p;

#define IGNORE_INDEX (-100)

// 16-byte global -> LDS copy. CDNA5 async-to-LDS (ASYNCcnt, no VGPR round
// trip); falls back to a synchronous copy if builtin is unavailable.
__device__ __forceinline__ void copy16_g2l(const __bf16* g, __bf16* l) {
#if __has_builtin(__builtin_amdgcn_global_load_async_to_lds_b128)
  __builtin_amdgcn_global_load_async_to_lds_b128(
      (v4i_as1p)(v4i*)g, (v4i_as3p)(v4i*)l, 0, 0);
#else
  *(v8bf*)l = *(const v8bf*)g;
#endif
}

__device__ __forceinline__ void wait_async_le2() {
#if __has_builtin(__builtin_amdgcn_global_load_async_to_lds_b128)
#if __has_builtin(__builtin_amdgcn_s_wait_asynccnt)
  __builtin_amdgcn_s_wait_asynccnt(2);
#else
  asm volatile("s_wait_asynccnt 0x2" ::: "memory");
#endif
#endif
}

// ---------------------------------------------------------------- converters
__global__ void cvt_f32_bf16(const float* __restrict__ src,
                             __bf16* __restrict__ dst, long n4) {
  long i = (long)blockIdx.x * blockDim.x + threadIdx.x;
  const long stride = (long)gridDim.x * blockDim.x;
  const float4* s4 = (const float4*)src;
  v4bf* d4 = (v4bf*)dst;
  for (; i < n4; i += stride) {
    float4 f = s4[i];
    v4bf h;
    h[0] = (__bf16)f.x; h[1] = (__bf16)f.y;
    h[2] = (__bf16)f.z; h[3] = (__bf16)f.w;
    d4[i] = h;
  }
}

// A-fragment load (16-bit A 16x32 ISA layout: lanes 0-15 hold K {0..7,16..23},
// lanes 16-31 hold K {8..15,24..31}); aofs = kk + hsel*8.
__device__ __forceinline__ v16bf loadA(const __bf16* row, int aofs) {
  v8bf lo = *(const v8bf*)(row + aofs);
  v8bf hi = *(const v8bf*)(row + aofs + 16);
  return __builtin_shufflevector(lo, hi,
      0,1,2,3,4,5,6,7,8,9,10,11,12,13,14,15);
}

// ------------------------------------------------------ main WMMA GEMM pass
// Block: 256 threads (8 waves). Tile: BM=256 rows x BN=64 vocab cols.
// Each wave: 32 rows x 64 cols = 8 accumulators (v8f each).
__global__ __launch_bounds__(256)
void lmhead_partial(const __bf16* __restrict__ Xb, const __bf16* __restrict__ Wb,
                    float* __restrict__ pmax, float* __restrict__ psum,
                    int Nrows, int V) {
  constexpr int H = 2048, BM = 256, BN = 64, BK = 32;
  constexpr int NT = H / BK;      // 64 K-tiles
  constexpr int D  = 4;           // async pipeline depth
  __shared__ __align__(32) __bf16 Bs[D][BN][BK];   // 4 x 4KB

  const int vb = blockIdx.x;
  const int v0 = vb * BN;
  const int r0blk = blockIdx.y * BM;
  const int tid = threadIdx.x;
  const int wave = tid >> 5;
  const int lane = tid & 31;
  const int lane16 = lane & 15;
  const int hsel = lane >> 4;      // half-wave K-split per ISA layout
  const int r0 = r0blk + wave * 32;

  const __bf16* arow0 = Xb + (size_t)(r0 + lane16) * H;
  const __bf16* arow1 = Xb + (size_t)(r0 + 16 + lane16) * H;

  // W-tile staging: 256 threads x 16B = 64 rows x 32 bf16 per tile.
  const int wrow = tid >> 2;            // 0..63  (vocab row within tile)
  const int wchk = (tid & 3) * 8;       // 0,8,16,24  (k chunk of 8 bf16)
  int gwr = v0 + wrow;
  if (gwr >= V) gwr = V - 1;            // clamp; masked in epilogue
  const __bf16* wsrc = Wb + (size_t)gwr * H + wchk;

  v8f acc[2][4] = {};

  // Prologue: stage first D-1 tiles, preload tile-0 A fragments.
#pragma unroll
  for (int d = 0; d < D - 1; ++d)
    copy16_g2l(wsrc + d * BK, &Bs[d][wrow][wchk]);
  v16bf a0 = loadA(arow0, hsel * 8);
  v16bf a1 = loadA(arow1, hsel * 8);

  for (int i = 0; i < NT; ++i) {
    // Async copies complete in-order: <=2 outstanding means tile i landed.
    wait_async_le2();
    __syncthreads();     // everyone has tile i; slot (i-1)%D free for reuse
    const int buf = i & (D - 1);

    // Preload next A fragments (consumed next iteration).
    v16bf na0 = a0, na1 = a1;
    if (i + 1 < NT) {
      const int naofs = (i + 1) * BK + hsel * 8;
      na0 = loadA(arow0, naofs);
      na1 = loadA(arow1, naofs);
    }

    // B 32x16 layout: lane n holds column v0+ct*16+n; half-wave h holds
    // K {8h..8h+7, 16+8h..16+8h+7} (matches A).
#pragma unroll
    for (int ct = 0; ct < 4; ++ct) {
      v16bf b = loadA(&Bs[buf][ct * 16 + lane16][0], hsel * 8);
      acc[0][ct] = __builtin_amdgcn_wmma_f32_16x16x32_bf16(
          false, a0, false, b, (short)0, acc[0][ct], false, false);
      acc[1][ct] = __builtin_amdgcn_wmma_f32_16x16x32_bf16(
          false, a1, false, b, (short)0, acc[1][ct], false, false);
    }

    // Stage tile i+D-1 into the slot just vacated (safe: barrier above).
    if (i + D - 1 < NT)
      copy16_g2l(wsrc + (i + D - 1) * BK, &Bs[(i + D - 1) & (D - 1)][wrow][wchk]);

    a0 = na0; a1 = na1;
  }

  // Epilogue: per-row max and sum(exp) over this block's 64 columns.
  // C layout: lanes 0-15 VGPR j = row M=j; lanes 16-31 VGPR j = row M=j+8.
  bool valid[4];
#pragma unroll
  for (int ct = 0; ct < 4; ++ct) valid[ct] = (v0 + ct * 16 + lane16) < V;

#pragma unroll
  for (int rt = 0; rt < 2; ++rt) {
    const int rowbase = r0 + rt * 16 + hsel * 8;
#pragma unroll
    for (int j = 0; j < 8; ++j) {
      float m = -INFINITY;
#pragma unroll
      for (int ct = 0; ct < 4; ++ct)
        if (valid[ct]) m = fmaxf(m, acc[rt][ct][j]);
#pragma unroll
      for (int ofs = 8; ofs >= 1; ofs >>= 1)
        m = fmaxf(m, __shfl_xor(m, ofs));      // half-wave row-max
      float s = 0.f;
#pragma unroll
      for (int ct = 0; ct < 4; ++ct)
        if (valid[ct]) s += __expf(acc[rt][ct][j] - m);
#pragma unroll
      for (int ofs = 8; ofs >= 1; ofs >>= 1)
        s += __shfl_xor(s, ofs);               // half-wave row-sum
      if (lane16 == 0) {
        const int row = rowbase + j;
        pmax[(size_t)vb * Nrows + row] = m;
        psum[(size_t)vb * Nrows + row] = s;
      }
    }
  }
}

// ----------------------------------------------- target logit (fp32, exact)
__global__ void target_logits(const float* __restrict__ x,
                              const float* __restrict__ w,
                              const int* __restrict__ y,
                              float* __restrict__ tgt, int Nrows, int V) {
  constexpr int H = 2048;
  const int wave = threadIdx.x >> 5, lane = threadIdx.x & 31;
  const int row = blockIdx.x * 8 + wave;
  if (row >= Nrows) return;
  int t = y[row];
  if (t < 0) t = 0;
  if (t >= V) t = V - 1;
  const float* xr = x + (size_t)row * H;
  const float* wr = w + (size_t)t * H;
  float s = 0.f;
  for (int k = lane; k < H; k += 32) s = fmaf(xr[k], wr[k], s);
#pragma unroll
  for (int ofs = 16; ofs >= 1; ofs >>= 1) s += __shfl_xor(s, ofs);
  if (lane == 0) tgt[row] = s;
}

// ---------------------------------------- merge partials -> per-row NLL
__global__ void row_nll(const float* __restrict__ pmax,
                        const float* __restrict__ psum,
                        const float* __restrict__ tgt,
                        const int* __restrict__ y,
                        float* __restrict__ nll, int Nrows, int nvb) {
  const int n = blockIdx.x * blockDim.x + threadIdx.x;
  if (n >= Nrows) return;
  float M = -INFINITY, S = 0.f;
  for (int vb = 0; vb < nvb; ++vb) {
    const float mb = pmax[(size_t)vb * Nrows + n];
    const float sb = psum[(size_t)vb * Nrows + n];
    const float nM = fmaxf(M, mb);
    if (nM == -INFINITY) continue;   // both empty
    S = S * __expf(M - nM) + sb * __expf(mb - nM);
    M = nM;
  }
  const float v = __logf(S) + M - tgt[n];
  nll[n] = (y[n] != IGNORE_INDEX) ? v : 0.f;
}

// -------------------------------------------------- final masked-mean reduce
__global__ void final_reduce(const float* __restrict__ nll,
                             const int* __restrict__ y,
                             float* __restrict__ out, int Nrows) {
  __shared__ float ssum[256];
  __shared__ int scnt[256];
  float s = 0.f;
  int c = 0;
  for (int n = threadIdx.x; n < Nrows; n += blockDim.x) {
    s += nll[n];
    c += (y[n] != IGNORE_INDEX) ? 1 : 0;
  }
  ssum[threadIdx.x] = s;
  scnt[threadIdx.x] = c;
  __syncthreads();
  for (int ofs = 128; ofs > 0; ofs >>= 1) {
    if (threadIdx.x < ofs) {
      ssum[threadIdx.x] += ssum[threadIdx.x + ofs];
      scnt[threadIdx.x] += scnt[threadIdx.x + ofs];
    }
    __syncthreads();
  }
  if (threadIdx.x == 0) {
    const int cv = scnt[0] > 0 ? scnt[0] : 1;
    out[0] = ssum[0] / (float)cv;
  }
}

// --------------------------------------------------------------------- host
extern "C" void kernel_launch(void* const* d_in, const int* in_sizes, int n_in,
                              void* d_out, int out_size, void* d_ws,
                              size_t ws_size, hipStream_t stream) {
  (void)in_sizes; (void)n_in; (void)out_size; (void)ws_size;
  constexpr int N = 4096, H = 2048, V = 50257;
  constexpr int BM = 256, BN = 64;
  constexpr int NVB = (V + BN - 1) / BN;   // 786

  const float* x = (const float*)d_in[0];
  const float* w = (const float*)d_in[1];
  const int*   y = (const int*)d_in[2];
  float* out = (float*)d_out;

  char* ws = (char*)d_ws;
  size_t off = 0;
  __bf16* xb = (__bf16*)(ws + off); off += (size_t)N * H * 2;
  __bf16* wb = (__bf16*)(ws + off); off += (size_t)V * H * 2;
  float* pmax = (float*)(ws + off); off += (size_t)NVB * N * 4;
  float* psum = (float*)(ws + off); off += (size_t)NVB * N * 4;
  float* tgt  = (float*)(ws + off); off += (size_t)N * 4;
  float* nll  = (float*)(ws + off); off += (size_t)N * 4;

  cvt_f32_bf16<<<1024, 256, 0, stream>>>(x, xb, (long)N * H / 4);
  cvt_f32_bf16<<<4096, 256, 0, stream>>>(w, wb, (long)V * H / 4);
  lmhead_partial<<<dim3(NVB, N / BM), 256, 0, stream>>>(xb, wb, pmax, psum, N, V);
  target_logits<<<N / 8, 256, 0, stream>>>(x, w, y, tgt, N, V);
  row_nll<<<(N + 255) / 256, 256, 0, stream>>>(pmax, psum, tgt, y, nll, N, NVB);
  final_reduce<<<1, 256, 0, stream>>>(nll, y, out, N);
}
